// BiLSTM_46608985096197
// MI455X (gfx1250) — matrix operand order repaired and checked by reading, not compiled, not run on hardware
//
#include <hip/hip_runtime.h>
#include <hip/hip_bf16.h>
#include <stdint.h>

// Problem constants (from reference)
#define S_LEN 8192
#define E_DIM 128
#define H_DIM 512
#define G4H   2048     // 4*H gate rows
#define MEET  4096     // S/2
#define NWG_SCAN 32    // persistent WGs per LSTM direction (64 KB bf16 W_hh slice each -> WGP$-resident)
#define SCAN_THREADS 256

typedef unsigned short v16u  __attribute__((ext_vector_type(16)));
typedef __bf16         v16bf __attribute__((ext_vector_type(16)));
typedef float          v8f   __attribute__((ext_vector_type(8)));

__device__ __forceinline__ unsigned short f2bf(float f) {
  unsigned int u = __float_as_uint(f);
  u += 0x7FFFu + ((u >> 16) & 1u);          // round-to-nearest-even
  return (unsigned short)(u >> 16);
}
__device__ __forceinline__ float sigf(float x) { return 1.f / (1.f + __expf(-x)); }

// ---------------------------------------------------------------------------
// 0) init: out[t] = b_lin, zero h double-buffer and barrier counters
// ---------------------------------------------------------------------------
__global__ void init_kernel(float* __restrict__ out, const float* __restrict__ b_lin,
                            float* __restrict__ hbuf, int* __restrict__ ctr) {
  int t = blockIdx.x * blockDim.x + threadIdx.x;
  if (t < S_LEN) out[t] = b_lin[0];
  if (t < 2 * 2 * H_DIM) hbuf[t] = 0.f;     // parity x lstm x H
  if (t < 64) ctr[t] = 0;
}

// ---------------------------------------------------------------------------
// 1) per-block column sums of gathered embeddings (blocked prefix-sum pass 1)
// ---------------------------------------------------------------------------
__global__ void colsum_kernel(const int* __restrict__ x, const float* __restrict__ emb,
                              float* __restrict__ bsum) {
  int e = threadIdx.x, b = blockIdx.x;
  float s = 0.f;
  for (int r = 0; r < 128; ++r) {
    int t = b * 128 + r;
    s += emb[(size_t)x[t] * E_DIM + e];
  }
  bsum[b * E_DIM + e] = s;
}

// 2) exclusive scan of the 64 block sums per column
__global__ void scanoff_kernel(const float* __restrict__ bsum, float* __restrict__ boff) {
  int e = threadIdx.x;
  float run = 0.f;
  for (int b = 0; b < 64; ++b) { boff[b * E_DIM + e] = run; run += bsum[b * E_DIM + e]; }
}

// 3) emit xx_fw = csum (bf16) and xx_bw = shifted/masked csum (bf16)
__global__ void writexx_kernel(const int* __restrict__ x, const float* __restrict__ emb,
                               const float* __restrict__ boff,
                               unsigned short* __restrict__ xxfw, unsigned short* __restrict__ xxbw) {
  int e = threadIdx.x, b = blockIdx.x;
  float run = boff[b * E_DIM + e];
  for (int r = 0; r < 128; ++r) {
    int t = b * 128 + r;
    run += emb[(size_t)x[t] * E_DIM + e];
    xxfw[(size_t)t * E_DIM + e] = f2bf(run);
    if (t < MEET) xxbw[(size_t)t * E_DIM + e] = 0;            // rows < meet are zero
    int u = t + 1;
    if (u >= MEET && u < S_LEN) xxbw[(size_t)u * E_DIM + e] = f2bf(run);  // row u = csum[u-1]
  }
}

// 4) weight conversion fp32 -> bf16, fused biases
__global__ void convert_kernel(const float* __restrict__ Wih1, const float* __restrict__ Whh1,
                               const float* __restrict__ bih1, const float* __restrict__ bhh1,
                               const float* __restrict__ Wih2, const float* __restrict__ Whh2,
                               const float* __restrict__ bih2, const float* __restrict__ bhh2,
                               unsigned short* __restrict__ Wih1b, unsigned short* __restrict__ Whh1b,
                               unsigned short* __restrict__ Wih2b, unsigned short* __restrict__ Whh2b,
                               float* __restrict__ bias1, float* __restrict__ bias2) {
  int i = blockIdx.x * blockDim.x + threadIdx.x;
  const int NIH = G4H * E_DIM;    // 262144
  const int NHH = G4H * H_DIM;    // 1048576
  if (i < NIH) { Wih1b[i] = f2bf(Wih1[i]); Wih2b[i] = f2bf(Wih2[i]); }
  if (i < NHH) { Whh1b[i] = f2bf(Whh1[i]); Whh2b[i] = f2bf(Whh2[i]); }
  if (i < G4H) { bias1[i] = bih1[i] + bhh1[i]; bias2[i] = bih2[i] + bhh2[i]; }
}

// ---------------------------------------------------------------------------
// 5) WMMA GEMM: pre[s][g] = sum_k xx[s][k] * W_ih[g][k] + (b_ih+b_hh)[g]
//    M=8192, N=2048, K=128, bf16 in, fp32 accumulate.
//    Each wave owns a 16x64 output strip (4 N-tiles): one A fragment is
//    reused across 4 v_wmma_f32_16x16x32_bf16 per k-step (16 WMMA/wave),
//    cutting A-side L2 traffic 4x. A/B use contiguous K per lane half; C/D per CDNA5 ISA 7.12.2.
// ---------------------------------------------------------------------------
#define NT_PER_WAVE 4
__global__ void gemm_pre_kernel(const unsigned short* __restrict__ xxfw,
                                const unsigned short* __restrict__ xxbw,
                                const unsigned short* __restrict__ Wih1b,
                                const unsigned short* __restrict__ Wih2b,
                                const float* __restrict__ bias1, const float* __restrict__ bias2,
                                float* __restrict__ pre1, float* __restrict__ pre2) {
  const unsigned short* xx  = (blockIdx.z == 0) ? xxfw  : xxbw;
  const unsigned short* wih = (blockIdx.z == 0) ? Wih1b : Wih2b;
  const float*          bia = (blockIdx.z == 0) ? bias1 : bias2;
  float*                pre = (blockIdx.z == 0) ? pre1  : pre2;

  int wid   = blockIdx.x * (blockDim.x >> 5) + (threadIdx.x >> 5);
  int lane  = threadIdx.x & 31;
  int ngrp  = wid & 31;                  // 2048 / (16*NT_PER_WAVE) = 32 N-groups
  int tileM = wid >> 5;                  // 8192/16 M-tiles
  int m0 = tileM << 4, n0 = ngrp << 6;   // 64-wide N strip

  // A (16x32 bf16): row = m0 + lane%16; lanes 0-15 hold K 0-15, lanes 16-31 hold K 16-31
  int arow  = m0 + (lane & 15);
  int abase = (lane < 16) ? 0 : 16;
  // B (32x16 bf16): col = n0 + 16*j + lane%16; lanes 0-15 hold K 0-15, lanes 16-31 K 16-31
  int bcol0 = n0 + (lane & 15);
  int bko   = (lane < 16) ? 0 : 16;

  v8f c[NT_PER_WAVE] = {};
#pragma unroll
  for (int kb = 0; kb < E_DIM; kb += 32) {
    v16u au;
    const unsigned short* ap = xx + (size_t)arow * E_DIM + kb + abase;
#pragma unroll
    for (int e = 0; e < 8; ++e) { au[e] = ap[e]; au[8 + e] = ap[8 + e]; }
    v16bf a = __builtin_bit_cast(v16bf, au);
#pragma unroll
    for (int j = 0; j < NT_PER_WAVE; ++j) {
      v16u bu;
      const unsigned short* bp = wih + (size_t)(bcol0 + 16 * j) * E_DIM + kb + bko;  // B[k][n]=W_ih[n][k]
#pragma unroll
      for (int e = 0; e < 16; ++e) bu[e] = bp[e];
      v16bf b = __builtin_bit_cast(v16bf, bu);
      // (neg_a, A, neg_b, B, c_mod, C, reuse_a, reuse_b)
      c[j] = __builtin_amdgcn_wmma_f32_16x16x32_bf16(false, a, false, b, (short)0, c[j], false, false);
    }
  }
  // C/D layout: VGPR r holds M = m0 + r + (lane>=16 ? 8 : 0), N = n0 + 16*j + lane%16
  int crow = m0 + ((lane < 16) ? 0 : 8);
#pragma unroll
  for (int j = 0; j < NT_PER_WAVE; ++j) {
    int nc = bcol0 + 16 * j;
    float bv = bia[nc];
#pragma unroll
    for (int r = 0; r < 8; ++r)
      pre[(size_t)(crow + r) * G4H + nc] = c[j][r] + bv;
  }
}

// ---------------------------------------------------------------------------
// 6) Persistent recurrent scan. 32 WGs per direction; WG w owns h[16w..16w+16)
//    -> 64 gate rows -> 64KB bf16 W_hh slice (WGP$-resident). Per step:
//    256 threads = 64 dots x 4 K-segments of 128 (128 MAC/thread), shfl reduce,
//    LSTM cell on 16 lanes, h exchanged via double-buffered global + counter
//    barrier. O=1 projection folded in via one atomicAdd per WG per step.
// ---------------------------------------------------------------------------
__global__ void scan_kernel(const float* __restrict__ pre1, const float* __restrict__ pre2,
                            const unsigned short* __restrict__ Whh1b,
                            const unsigned short* __restrict__ Whh2b,
                            const float* __restrict__ Wlin,
                            float* __restrict__ hbuf, int* __restrict__ ctrbase,
                            float* __restrict__ out) {
  int lstm = blockIdx.x / NWG_SCAN;
  int w    = blockIdx.x % NWG_SCAN;
  int tid  = threadIdx.x;
  const float* pre          = (lstm == 0) ? pre1  : pre2;
  const unsigned short* whh = (lstm == 0) ? Whh1b : Whh2b;
  int* ctr = ctrbase + lstm * 16;            // 64B apart

  __shared__ float h_lds[H_DIM];
  __shared__ float gates[64];
  __shared__ float c_lds[16];

  for (int i = tid; i < H_DIM; i += SCAN_THREADS) h_lds[i] = 0.f;  // h0 = 0
  if (tid < 16) c_lds[tid] = 0.f;                                   // c0 = 0
  __syncthreads();

  int d = tid >> 2, seg = tid & 3;           // dot id, K segment
  int gt = d >> 4;                           // gate type (i,f,g,o)
  int jj = w * 16 + (d & 15);                // hidden index
  int r  = gt * H_DIM + jj;                  // gate row in W_hh / pre
  const unsigned int* wrow = (const unsigned int*)(whh + (size_t)r * H_DIM + seg * 128);
  const float* hseg = h_lds + seg * 128;
  float wlv = (tid < 16) ? Wlin[lstm * H_DIM + w * 16 + tid] : 0.f;

  for (int t = 0; t < S_LEN; ++t) {
    float acc = 0.f;
#pragma unroll 8
    for (int k = 0; k < 64; ++k) {           // 128 bf16 MACs, packed-unpack from uint
      unsigned int u = wrow[k];
      acc += __uint_as_float(u << 16)          * hseg[2 * k]
           + __uint_as_float(u & 0xFFFF0000u) * hseg[2 * k + 1];
    }
    acc += __shfl_xor(acc, 1, 32);           // reduce 4 K-segments (consecutive lanes)
    acc += __shfl_xor(acc, 2, 32);
    if (seg == 0) gates[d] = acc + pre[(size_t)t * G4H + r];
    __syncthreads();

    if (tid < 16) {
      float gi = sigf(gates[tid]);
      float gf = sigf(gates[16 + tid]);
      float gg = tanhf(gates[32 + tid]);
      float go = sigf(gates[48 + tid]);
      float cc = gf * c_lds[tid] + gi * gg;
      c_lds[tid] = cc;
      float hh = go * tanhf(cc);
      int parity = t & 1;                    // double buffer: no read/write overlap across steps
      __hip_atomic_store(&hbuf[parity * (2 * H_DIM) + lstm * H_DIM + w * 16 + tid], hh,
                         __ATOMIC_RELAXED, __HIP_MEMORY_SCOPE_AGENT);
      float p = hh * wlv;                    // fold O=1 projection
      p += __shfl_xor(p, 1, 32); p += __shfl_xor(p, 2, 32);
      p += __shfl_xor(p, 4, 32); p += __shfl_xor(p, 8, 32);
      if (tid == 0) {
        int pos = (lstm == 0) ? t : (S_LEN - 1 - t);   // bw pairs with reversed index
        atomicAdd(&out[pos], p);
      }
    }
    __threadfence();                         // release h stores
    __syncthreads();
    if (tid == 0) {                          // device barrier across the 32 WGs of this direction
      atomicAdd(ctr, 1);
      int target = NWG_SCAN * (t + 1);
      while (__hip_atomic_load(ctr, __ATOMIC_RELAXED, __HIP_MEMORY_SCOPE_AGENT) < target) { }
      __threadfence();                       // acquire
    }
    __syncthreads();
    int parity = t & 1;
    for (int i = tid; i < H_DIM; i += SCAN_THREADS)
      h_lds[i] = __hip_atomic_load(&hbuf[parity * (2 * H_DIM) + lstm * H_DIM + i],
                                   __ATOMIC_RELAXED, __HIP_MEMORY_SCOPE_AGENT);
    __syncthreads();
  }
}

// ---------------------------------------------------------------------------
extern "C" void kernel_launch(void* const* d_in, const int* in_sizes, int n_in,
                              void* d_out, int out_size, void* d_ws, size_t ws_size,
                              hipStream_t stream) {
  (void)in_sizes; (void)n_in; (void)out_size; (void)ws_size;
  const int*   x     = (const int*)  d_in[0];
  const float* emb   = (const float*)d_in[1];
  const float* Wih1  = (const float*)d_in[2];
  const float* Whh1  = (const float*)d_in[3];
  const float* bih1  = (const float*)d_in[4];
  const float* bhh1  = (const float*)d_in[5];
  const float* Wih2  = (const float*)d_in[6];
  const float* Whh2  = (const float*)d_in[7];
  const float* bih2  = (const float*)d_in[8];
  const float* bhh2  = (const float*)d_in[9];
  const float* Wlin  = (const float*)d_in[10];
  const float* blin  = (const float*)d_in[11];
  float* out = (float*)d_out;

  char* base = (char*)d_ws;
  size_t off = 0;
  auto alloc = [&](size_t bytes) -> char* {
    char* p = base + off;
    off = (off + bytes + 255) & ~(size_t)255;
    return p;
  };
  unsigned short* xxfw  = (unsigned short*)alloc((size_t)S_LEN * E_DIM * 2);
  unsigned short* xxbw  = (unsigned short*)alloc((size_t)S_LEN * E_DIM * 2);
  unsigned short* Wih1b = (unsigned short*)alloc((size_t)G4H * E_DIM * 2);
  unsigned short* Wih2b = (unsigned short*)alloc((size_t)G4H * E_DIM * 2);
  unsigned short* Whh1b = (unsigned short*)alloc((size_t)G4H * H_DIM * 2);
  unsigned short* Whh2b = (unsigned short*)alloc((size_t)G4H * H_DIM * 2);
  float* bias1 = (float*)alloc((size_t)G4H * 4);
  float* bias2 = (float*)alloc((size_t)G4H * 4);
  float* bsum  = (float*)alloc((size_t)64 * E_DIM * 4);
  float* boff  = (float*)alloc((size_t)64 * E_DIM * 4);
  float* hbuf  = (float*)alloc((size_t)2 * 2 * H_DIM * 4);
  int*   ctr   = (int*)  alloc(256);
  float* pre1  = (float*)alloc((size_t)S_LEN * G4H * 4);   // 64 MB
  float* pre2  = (float*)alloc((size_t)S_LEN * G4H * 4);   // 64 MB

  init_kernel<<<32, 256, 0, stream>>>(out, blin, hbuf, ctr);
  colsum_kernel<<<64, 128, 0, stream>>>(x, emb, bsum);
  scanoff_kernel<<<1, 128, 0, stream>>>(bsum, boff);
  writexx_kernel<<<64, 128, 0, stream>>>(x, emb, boff, xxfw, xxbw);
  convert_kernel<<<4096, 256, 0, stream>>>(Wih1, Whh1, bih1, bhh1,
                                           Wih2, Whh2, bih2, bhh2,
                                           Wih1b, Whh1b, Wih2b, Whh2b, bias1, bias2);
  // 512 M-tiles * 32 N-groups = 16384 waves per direction; 8 waves/block
  gemm_pre_kernel<<<dim3(2048, 1, 2), 256, 0, stream>>>(xxfw, xxbw, Wih1b, Wih2b,
                                                        bias1, bias2, pre1, pre2);
  scan_kernel<<<2 * NWG_SCAN, SCAN_THREADS, 0, stream>>>(pre1, pre2, Whh1b, Whh2b,
                                                         Wlin, hbuf, ctr, out);
}
